// MiniGPT_28484223107450
// MI455X (gfx1250) — compile-verified, matrix operand order checked by reading, not run on hardware
//
#include <hip/hip_runtime.h>
#include <hip/hip_bf16.h>
#include <math.h>

// MiniGPT forward for MI455X (gfx1250, wave32, WMMA + TDM).
// All GEMMs run on v_wmma_f32_16x16x32_f16 (f16 inputs, fp32 accumulate).
// A tiles (f16 activations) are DMA'd into LDS by the Tensor Data Mover;
// fp32 weights are converted to f16 while staging B tiles into LDS.

typedef __attribute__((ext_vector_type(16))) _Float16 v16h;
typedef __attribute__((ext_vector_type(8)))  _Float16 v8h;
typedef __attribute__((ext_vector_type(8)))  float    v8f;
typedef __attribute__((ext_vector_type(4)))  unsigned int u32x4;
typedef __attribute__((ext_vector_type(8)))  int      i32x8;
typedef __attribute__((ext_vector_type(4)))  int      i32x4;

#define B_   4
#define T_   1024
#define E_   1024
#define H_   16
#define HD_  64
#define L_   8
#define V_   32000
#define FF_  4096
#define MTOK (B_ * T_)          // 4096 token rows

// Epilogue modes for the generic WMMA GEMM
#define EP_F32_BIAS_RES 0       // out_f32 = acc + bias[n] + res[m,n]   (residual update)
#define EP_F16_QKV      1       // out_f16 relayout to [B,H,T,HD], no bias
#define EP_F16_GELU     2       // out_f16 = gelu(acc + bias[n])
#define EP_F32_BIAS     3       // out_f32 = acc + bias[n]              (logits)

#if defined(__has_builtin)
#if __has_builtin(__builtin_amdgcn_tensor_load_to_lds)
#define USE_TDM 1
#endif
#if __has_builtin(__builtin_amdgcn_s_wait_tensorcnt)
#define HAVE_WAIT_TENSORCNT 1
#endif
#endif

static __device__ __forceinline__ void wait_tensorcnt0() {
#if defined(HAVE_WAIT_TENSORCNT)
  __builtin_amdgcn_s_wait_tensorcnt(0);
#else
  asm volatile("s_wait_tensorcnt 0x0" ::: "memory");
#endif
}

// ---------------------------------------------------------------------------
// Fragment load following CDNA5 ISA 7.12.2 16-bit A-matrix layout:
// lane L: row m = L%16, half = L/16; VGPR v holds elements at
// k = (v/4)*16 + half*8 + (v%4)*2  and k+1  -> two contiguous 8xf16 chunks.
// Works for A frags and (with [n][k]-transposed storage) B frags.
// ---------------------------------------------------------------------------
static __device__ __forceinline__ v16h load_frag(const _Float16* p, int ld) {
  const int lane = threadIdx.x & 31;
  const int m    = lane & 15;
  const int half = lane >> 4;
  const _Float16* p0 = p + m * ld + half * 8;
  v8h lo = *reinterpret_cast<const v8h*>(p0);
  v8h hi = *reinterpret_cast<const v8h*>(p0 + 16);
  v16h r;
#pragma unroll
  for (int i = 0; i < 8; ++i) { r[i] = lo[i]; r[8 + i] = hi[i]; }
  return r;
}

static __device__ __forceinline__ v8f wmma16(v16h a, v16h b, v8f c) {
  // (neg_a, A, neg_b, B, c_mod, C, reuse_a, reuse_b)
  return __builtin_amdgcn_wmma_f32_16x16x32_f16(false, a, false, b, (short)0, c,
                                                false, false);
}

// ---------------------------------------------------------------------------
// TDM: DMA a 2D tile [128 rows x 32 halves] from a row-major f16 tensor with
// row stride K into LDS (contiguous [row][k] layout). Issued by one wave;
// caller must wait_tensorcnt0() + barrier before consuming.
// D# layout per CDNA5 ISA ch.8 (group0: count/lds/global/type, group1: dims).
// ---------------------------------------------------------------------------
#if defined(USE_TDM)
static __device__ __forceinline__ void tdm_load_a_tile(const _Float16* gsrc,
                                                       _Float16* lds_dst,
                                                       int K, int M_total) {
  const unsigned long long ga = (unsigned long long)(const void*)gsrc;
  const unsigned lds = (unsigned)(unsigned long long)(const void*)lds_dst; // addr[31:0] = LDS offset
  u32x4 g0;
  g0[0] = 1u;                                    // count=1 (valid), user mode
  g0[1] = lds;                                   // lds_addr
  g0[2] = (unsigned)ga;                          // global_addr[31:0]
  g0[3] = (unsigned)(ga >> 32) | (2u << 30);     // global_addr[56:32] | type=2 ("image")
  i32x8 g1;
  g1[0] = 1 << 16;                               // data_size = 1 -> 2 bytes
  g1[1] = (int)((K & 0xFFFF) << 16);             // tensor_dim0[15:0]
  g1[2] = (int)((K >> 16) | ((M_total & 0xFFFF) << 16));   // dim0 hi | tensor_dim1 lo
  g1[3] = (int)((M_total >> 16) | (32 << 16));   // tensor_dim1 hi | tile_dim0 = 32
  g1[4] = 128;                                   // tile_dim1 = 128, tile_dim2 = 0
  g1[5] = K;                                     // tensor_dim0_stride[31:0]
  g1[6] = 0;                                     // stride0 hi | stride1 lo
  g1[7] = 0;
  i32x4 z4; z4[0] = 0; z4[1] = 0; z4[2] = 0; z4[3] = 0;
#if __clang_major__ >= 23
  i32x8 z8; z8[0]=0; z8[1]=0; z8[2]=0; z8[3]=0; z8[4]=0; z8[5]=0; z8[6]=0; z8[7]=0;
  __builtin_amdgcn_tensor_load_to_lds(g0, g1, z4, z4, z8, 0);
#else
  __builtin_amdgcn_tensor_load_to_lds(g0, g1, z4, z4, 0);
#endif
}
#endif

// ---------------------------------------------------------------------------
// Embedding: h[b*T+t, :] = tok_emb[x[b,t], :] + pos_emb[t, :]
// ---------------------------------------------------------------------------
__launch_bounds__(256)
__global__ void embed_kernel(const int* __restrict__ x, const float* __restrict__ tok,
                             const float* __restrict__ pos, float* __restrict__ h) {
  const int row = blockIdx.x;            // 0..4095
  const int t   = threadIdx.x;           // 256 threads * float4 = 1024
  const int tid = x[row];
  const int tt  = row & (T_ - 1);
  float4 a = reinterpret_cast<const float4*>(tok + (size_t)tid * E_)[t];
  float4 p = reinterpret_cast<const float4*>(pos + (size_t)tt * E_)[t];
  float4 o;
  o.x = a.x + p.x; o.y = a.y + p.y; o.z = a.z + p.z; o.w = a.w + p.w;
  reinterpret_cast<float4*>(h + (size_t)row * E_)[t] = o;
}

// ---------------------------------------------------------------------------
// LayerNorm (fp32 in) -> f16 out. One block (256 threads) per row of 1024.
// ---------------------------------------------------------------------------
__launch_bounds__(256)
__global__ void ln_f16_kernel(const float* __restrict__ x, const float* __restrict__ g,
                              const float* __restrict__ b, _Float16* __restrict__ out) {
  __shared__ float red[256];
  const int row = blockIdx.x;
  const int t   = threadIdx.x;
  float4 vv = reinterpret_cast<const float4*>(x + (size_t)row * E_)[t];

  red[t] = vv.x + vv.y + vv.z + vv.w;
  __syncthreads();
#pragma unroll
  for (int o = 128; o > 0; o >>= 1) {
    if (t < o) red[t] += red[t + o];
    __syncthreads();
  }
  const float mean = red[0] * (1.0f / E_);
  __syncthreads();

  const float d0 = vv.x - mean, d1 = vv.y - mean, d2 = vv.z - mean, d3 = vv.w - mean;
  red[t] = d0 * d0 + d1 * d1 + d2 * d2 + d3 * d3;
  __syncthreads();
#pragma unroll
  for (int o = 128; o > 0; o >>= 1) {
    if (t < o) red[t] += red[t + o];
    __syncthreads();
  }
  const float rstd = rsqrtf(red[0] * (1.0f / E_) + 1e-5f);

  const int c = t * 4;
  _Float16* op = out + (size_t)row * E_ + c;
  op[0] = (_Float16)(d0 * rstd * g[c + 0] + b[c + 0]);
  op[1] = (_Float16)(d1 * rstd * g[c + 1] + b[c + 1]);
  op[2] = (_Float16)(d2 * rstd * g[c + 2] + b[c + 2]);
  op[3] = (_Float16)(d3 * rstd * g[c + 3] + b[c + 3]);
}

// ---------------------------------------------------------------------------
// Generic WMMA GEMM:  C[M,N] = A_f16[M,K] * B_f32[K,N] (+epilogue)
//  - 128x128 block tile, 256 threads = 8 waves, each wave 64x32 (4x2 WMMA tiles)
//  - A tile DMA'd to LDS by TDM (wave 0 issues, overlaps with B staging)
//  - B: each thread owns 16 CONTIGUOUS fp32 elements (also true in gather
//    mode since 16-aligned chunks never cross a 64-wide head): 4x float4
//    loads in one clause -> cvt -> transposed [n][k] LDS stores.
//  - GATHER: B[k][n] = W[h=n/64][k][d=n%64]  (layout [H,E,HD])
// ---------------------------------------------------------------------------
template <int EPI, bool GATHER>
__launch_bounds__(256)
__global__ void wmma_gemm(const _Float16* __restrict__ A, const float* __restrict__ Bm,
                          const float* __restrict__ bias, const float* __restrict__ res,
                          float* __restrict__ oF32, _Float16* __restrict__ oF16,
                          int M, int N, int K) {
  __shared__ __align__(16) _Float16 sA[128 * 32];   // [m][k]
  __shared__ __align__(16) _Float16 sB[128 * 32];   // [n][k]  (transposed)

  const int m0   = blockIdx.x * 128;
  const int n0   = blockIdx.y * 128;
  const int tid  = threadIdx.x;
  const int wave = tid >> 5;
  const int wm   = (wave >> 2) * 64;   // wave row offset in tile
  const int wn   = (wave & 3) * 32;    // wave col offset in tile

  // Per-thread B-staging geometry: 16 contiguous n at fixed k.
  const int kl   = tid >> 3;           // 0..31
  const int nb   = (tid & 7) * 16;     // 0..112
  const int nAbs = n0 + nb;
  // Contiguous base for this thread's 16 elements at k-row (k0+kl):
  //   normal: Bm + (k0+kl)*N + nAbs          (advance by 32*N per k-step)
  //   gather: Bm + (nAbs/64)*K*64 + (k0+kl)*64 + nAbs%64   (advance by 32*64)
  const size_t bRowStep = GATHER ? (size_t)32 * 64 : (size_t)32 * N;
  const float* bSrc = GATHER
      ? Bm + ((size_t)(nAbs >> 6) * (size_t)K + (size_t)kl) * 64 + (nAbs & 63)
      : Bm + (size_t)kl * (size_t)N + nAbs;

  v8f acc[4][2];
#pragma unroll
  for (int i = 0; i < 4; ++i)
#pragma unroll
    for (int j = 0; j < 2; ++j)
#pragma unroll
      for (int r = 0; r < 8; ++r) acc[i][j][r] = 0.0f;

  for (int k0 = 0; k0 < K; k0 += 32, bSrc += bRowStep) {
    __syncthreads();
    // ---- stage A tile: 128x32 f16 via Tensor Data Mover (wave 0) ----
#if defined(USE_TDM)
    if (wave == 0) {
      tdm_load_a_tile(A + (size_t)m0 * K + k0, sA, K, M);
    }
#else
    {
      const int ml = tid >> 1;
      const int ko = (tid & 1) * 16;
      const _Float16* srcA = A + (size_t)(m0 + ml) * K + k0 + ko;
      *reinterpret_cast<v8h*>(&sA[ml * 32 + ko])     = *reinterpret_cast<const v8h*>(srcA);
      *reinterpret_cast<v8h*>(&sA[ml * 32 + ko + 8]) = *reinterpret_cast<const v8h*>(srcA + 8);
    }
#endif
    // ---- stage B tile: 16 contiguous fp32 per thread, one load clause ----
    {
      if (k0 + 32 < K)  // speculative prefetch of next K-slice (global_prefetch_b8)
        __builtin_prefetch(bSrc + bRowStep, 0, 1);
      const float4* bv = reinterpret_cast<const float4*>(bSrc);
      float4 c0 = bv[0];
      float4 c1 = bv[1];
      float4 c2 = bv[2];
      float4 c3 = bv[3];
      const float fv[16] = { c0.x, c0.y, c0.z, c0.w, c1.x, c1.y, c1.z, c1.w,
                             c2.x, c2.y, c2.z, c2.w, c3.x, c3.y, c3.z, c3.w };
#pragma unroll
      for (int j = 0; j < 16; ++j) sB[(nb + j) * 32 + kl] = (_Float16)fv[j];
    }
#if defined(USE_TDM)
    if (wave == 0) wait_tensorcnt0();   // TENSORcnt is per-wave; barrier publishes
#endif
    __syncthreads();

    v16h af[4];
#pragma unroll
    for (int i = 0; i < 4; ++i) af[i] = load_frag(&sA[(wm + i * 16) * 32], 32);
#pragma unroll
    for (int j = 0; j < 2; ++j) {
      v16h bf = load_frag(&sB[(wn + j * 16) * 32], 32);
#pragma unroll
      for (int i = 0; i < 4; ++i) acc[i][j] = wmma16(af[i], bf, acc[i][j]);
    }
  }

  // ---- epilogue ----
  const int lane = tid & 31, nl = lane & 15, half = lane >> 4;
#pragma unroll
  for (int i = 0; i < 4; ++i)
#pragma unroll
    for (int j = 0; j < 2; ++j)
#pragma unroll
      for (int r = 0; r < 8; ++r) {
        const int m = m0 + wm + i * 16 + r + 8 * half;   // C layout: row = vgpr + 8*half
        const int n = n0 + wn + j * 16 + nl;             //           col = lane%16
        const float val = acc[i][j][r];
        if constexpr (EPI == EP_F32_BIAS_RES) {
          const size_t o = (size_t)m * N + n;
          oF32[o] = val + bias[n] + res[o];
        } else if constexpr (EPI == EP_F16_QKV) {
          const int bb = m >> 10, tt = m & (T_ - 1);
          const size_t o = ((size_t)(bb * H_ + (n >> 6)) * T_ + tt) * HD_ + (n & 63);
          oF16[o] = (_Float16)val;
        } else if constexpr (EPI == EP_F16_GELU) {
          const float xv = val + bias[n];
          const float gl = 0.5f * xv * (1.0f + erff(xv * 0.70710678f));
          oF16[(size_t)m * N + n] = (_Float16)gl;
        } else {  // EP_F32_BIAS
          oF32[(size_t)m * N + n] = val + bias[n];
        }
      }
}

// ---------------------------------------------------------------------------
// Flash attention (causal), q/k/v in f16 [B,H,T,HD].
// Grid: (B*H, T/64), 128 threads = 4 waves; wave owns 16 query rows.
// S = Q K^T via WMMA (K frags loaded straight from global: B[k=d][n]=K[n][d]
// matches the A-frag pattern); all 4 frags of a K-step are loaded before the
// WMMAs so the loads pipeline. Online softmax with 16-lane xor reductions.
// P.V via WMMA with V^T staged in LDS. SCALE = E^-0.5 = 1/32 (per reference).
// ---------------------------------------------------------------------------
__launch_bounds__(128)
__global__ void attn_kernel(const _Float16* __restrict__ q, const _Float16* __restrict__ k,
                            const _Float16* __restrict__ v, _Float16* __restrict__ aout) {
  __shared__ __align__(16) _Float16 VT[64 * 64];     // [d][s_local]
  __shared__ __align__(16) _Float16 P[4][16 * 64];   // per-wave probs [m][s_local]

  const int bh   = blockIdx.x;
  const int bb   = bh >> 4, hh = bh & 15;
  const int qt0  = blockIdx.y * 64;
  const int tid  = threadIdx.x;
  const int wave = tid >> 5, lane = tid & 31;
  const int nl   = lane & 15, half = lane >> 4;
  const int qrow0 = qt0 + wave * 16;
  const size_t base = (size_t)bh * T_ * HD_;
  const float scale = 0.03125f;  // 1024^-0.5

  v16h qf[2];
#pragma unroll
  for (int p = 0; p < 2; ++p)
    qf[p] = load_frag(q + base + (size_t)qrow0 * HD_ + p * 32, HD_);

  v8f o[4];
  float mrow[8], lrow[8];
#pragma unroll
  for (int tn = 0; tn < 4; ++tn)
#pragma unroll
    for (int r = 0; r < 8; ++r) o[tn][r] = 0.0f;
#pragma unroll
  for (int r = 0; r < 8; ++r) { mrow[r] = -1e30f; lrow[r] = 0.0f; }

  for (int s0 = 0; s0 <= qt0; s0 += 64) {
    __syncthreads();
    // ---- stage V^T tile (64 keys x 64 dims) ----
    {
      const int sr = tid >> 1, dco = (tid & 1) * 32;
      const v8h* vr = reinterpret_cast<const v8h*>(v + base + (size_t)(s0 + sr) * HD_ + dco);
#pragma unroll
      for (int c = 0; c < 4; ++c) {
        v8h ch = vr[c];
#pragma unroll
        for (int j = 0; j < 8; ++j) VT[(dco + c * 8 + j) * 64 + sr] = ch[j];
      }
    }
    __syncthreads();

    // ---- S = Q K^T (16x64 per wave, 8 WMMAs; frag loads batched per K-step) ----
    v8f st[4];
    v16h kf[4];
#pragma unroll
    for (int tn = 0; tn < 4; ++tn)
      kf[tn] = load_frag(k + base + (size_t)(s0 + tn * 16) * HD_, HD_);
#pragma unroll
    for (int tn = 0; tn < 4; ++tn) {
      v8f z;
#pragma unroll
      for (int r = 0; r < 8; ++r) z[r] = 0.0f;
      st[tn] = wmma16(qf[0], kf[tn], z);
    }
#pragma unroll
    for (int tn = 0; tn < 4; ++tn)
      kf[tn] = load_frag(k + base + (size_t)(s0 + tn * 16) * HD_ + 32, HD_);
#pragma unroll
    for (int tn = 0; tn < 4; ++tn)
      st[tn] = wmma16(qf[1], kf[tn], st[tn]);

    // ---- causal mask + online softmax (row stats per C-matrix row) ----
#pragma unroll
    for (int r = 0; r < 8; ++r) {
      const int qr = qrow0 + r + 8 * half;
      float sv[4];
      float mx = -1e30f;
#pragma unroll
      for (int tn = 0; tn < 4; ++tn) {
        const int kc = s0 + tn * 16 + nl;
        const float x = (kc <= qr) ? st[tn][r] * scale : -1e30f;
        sv[tn] = x;
        mx = fmaxf(mx, x);
      }
#pragma unroll
      for (int m = 1; m < 16; m <<= 1) mx = fmaxf(mx, __shfl_xor(mx, m, 32));
      const float nm    = fmaxf(mrow[r], mx);
      const float alpha = __expf(mrow[r] - nm);
      float ps = 0.0f;
#pragma unroll
      for (int tn = 0; tn < 4; ++tn) {
        const float pr = __expf(sv[tn] - nm);
        ps += pr;
        P[wave][(r + 8 * half) * 64 + tn * 16 + nl] = (_Float16)pr;
      }
#pragma unroll
      for (int m = 1; m < 16; m <<= 1) ps += __shfl_xor(ps, m, 32);
      lrow[r] = lrow[r] * alpha + ps;
      mrow[r] = nm;
#pragma unroll
      for (int tn = 0; tn < 4; ++tn) o[tn][r] *= alpha;
    }
    __syncthreads();

    // ---- O += P * V  (A frags from P in LDS, B frags from V^T; batched) ----
#pragma unroll
    for (int p = 0; p < 2; ++p) {
      v16h pf = load_frag(&P[wave][0] + p * 32, 64);
      v16h vf[4];
#pragma unroll
      for (int tn = 0; tn < 4; ++tn)
        vf[tn] = load_frag(&VT[(tn * 16) * 64] + p * 32, 64);
#pragma unroll
      for (int tn = 0; tn < 4; ++tn)
        o[tn] = wmma16(pf, vf[tn], o[tn]);
    }
  }

  // ---- normalize and write concat-head layout [B,T,E] ----
#pragma unroll
  for (int r = 0; r < 8; ++r) {
    const int qr = qrow0 + r + 8 * half;
    const float inv = 1.0f / lrow[r];
#pragma unroll
    for (int tn = 0; tn < 4; ++tn) {
      const size_t oi = ((size_t)bb * T_ + qr) * E_ + hh * HD_ + tn * 16 + nl;
      aout[oi] = (_Float16)(o[tn][r] * inv);
    }
  }
}

// ---------------------------------------------------------------------------
// Host-side orchestration
// ---------------------------------------------------------------------------
extern "C" void kernel_launch(void* const* d_in, const int* in_sizes, int n_in,
                              void* d_out, int out_size, void* d_ws, size_t ws_size,
                              hipStream_t stream) {
  (void)in_sizes; (void)n_in; (void)out_size; (void)ws_size;

  const int*   x     = (const int*)  d_in[0];
  const float* tok   = (const float*)d_in[1];
  const float* pos   = (const float*)d_in[2];
  const float* Wq    = (const float*)d_in[3];
  const float* Wk    = (const float*)d_in[4];
  const float* Wv    = (const float*)d_in[5];
  const float* Wproj = (const float*)d_in[6];
  const float* bproj = (const float*)d_in[7];
  const float* ln1g  = (const float*)d_in[8];
  const float* ln1b  = (const float*)d_in[9];
  const float* ln2g  = (const float*)d_in[10];
  const float* ln2b  = (const float*)d_in[11];
  const float* W1    = (const float*)d_in[12];
  const float* b1    = (const float*)d_in[13];
  const float* W2    = (const float*)d_in[14];
  const float* b2    = (const float*)d_in[15];
  const float* lnfg  = (const float*)d_in[16];
  const float* lnfb  = (const float*)d_in[17];
  const float* Wout  = (const float*)d_in[18];
  const float* bout  = (const float*)d_in[19];
  float* logits = (float*)d_out;

  // Workspace layout (~92 MB)
  char* w = (char*)d_ws;
  float*    h    = (float*)w;    w += (size_t)MTOK * E_  * sizeof(float);
  _Float16* xn   = (_Float16*)w; w += (size_t)MTOK * E_  * sizeof(_Float16);
  _Float16* q16  = (_Float16*)w; w += (size_t)MTOK * E_  * sizeof(_Float16);
  _Float16* k16  = (_Float16*)w; w += (size_t)MTOK * E_  * sizeof(_Float16);
  _Float16* v16  = (_Float16*)w; w += (size_t)MTOK * E_  * sizeof(_Float16);
  _Float16* a16  = (_Float16*)w; w += (size_t)MTOK * E_  * sizeof(_Float16);
  _Float16* ff16 = (_Float16*)w; w += (size_t)MTOK * FF_ * sizeof(_Float16);

  embed_kernel<<<MTOK, 256, 0, stream>>>(x, tok, pos, h);

  const dim3 gE(MTOK / 128, E_ / 128);    // 32 x 8
  const dim3 gF(MTOK / 128, FF_ / 128);   // 32 x 32
  const dim3 gA(B_ * H_, T_ / 64);        // 64 x 16

  for (int l = 0; l < L_; ++l) {
    const float* wq = Wq    + (size_t)l * H_ * E_ * HD_;
    const float* wk = Wk    + (size_t)l * H_ * E_ * HD_;
    const float* wv = Wv    + (size_t)l * H_ * E_ * HD_;
    const float* wp = Wproj + (size_t)l * E_ * E_;
    const float* bp = bproj + (size_t)l * E_;
    const float* w1 = W1    + (size_t)l * E_ * FF_;
    const float* bb1 = b1   + (size_t)l * FF_;
    const float* w2 = W2    + (size_t)l * FF_ * E_;
    const float* bb2 = b2   + (size_t)l * E_;

    ln_f16_kernel<<<MTOK, 256, 0, stream>>>(h, ln1g + (size_t)l * E_, ln1b + (size_t)l * E_, xn);

    wmma_gemm<EP_F16_QKV, true><<<gE, 256, 0, stream>>>(xn, wq, nullptr, nullptr,
                                                        nullptr, q16, MTOK, E_, E_);
    wmma_gemm<EP_F16_QKV, true><<<gE, 256, 0, stream>>>(xn, wk, nullptr, nullptr,
                                                        nullptr, k16, MTOK, E_, E_);
    wmma_gemm<EP_F16_QKV, true><<<gE, 256, 0, stream>>>(xn, wv, nullptr, nullptr,
                                                        nullptr, v16, MTOK, E_, E_);

    attn_kernel<<<gA, 128, 0, stream>>>(q16, k16, v16, a16);

    wmma_gemm<EP_F32_BIAS_RES, false><<<gE, 256, 0, stream>>>(a16, wp, bp, h,
                                                              h, nullptr, MTOK, E_, E_);

    ln_f16_kernel<<<MTOK, 256, 0, stream>>>(h, ln2g + (size_t)l * E_, ln2b + (size_t)l * E_, xn);

    wmma_gemm<EP_F16_GELU, false><<<gF, 256, 0, stream>>>(xn, w1, bb1, nullptr,
                                                          nullptr, ff16, MTOK, FF_, E_);
    wmma_gemm<EP_F32_BIAS_RES, false><<<gE, 256, 0, stream>>>(ff16, w2, bb2, h,
                                                              h, nullptr, MTOK, E_, FF_);
  }

  ln_f16_kernel<<<MTOK, 256, 0, stream>>>(h, lnfg, lnfb, xn);

  const dim3 gV(MTOK / 128, V_ / 128);    // 32 x 250
  wmma_gemm<EP_F32_BIAS, false><<<gV, 256, 0, stream>>>(xn, Wout, bout, nullptr,
                                                        logits, nullptr, MTOK, V_, E_);
}
